// GATModel_40123584479256
// MI455X (gfx1250) — matrix operand order repaired and checked by reading, not compiled, not run on hardware
//
#include <hip/hip_runtime.h>
#include <hip/hip_bf16.h>
#include <math.h>

#define HID   256
#define HEADS 4
#define HD    64
#define BN_EPS 1e-5f

typedef __bf16 bf16_t;
typedef __attribute__((ext_vector_type(16))) bf16_t bf16x16;
typedef __attribute__((ext_vector_type(8)))  bf16_t bf16x8;
typedef __attribute__((ext_vector_type(8)))  float  f32x8;

// ---------------------------------------------------------------------------
// fp32 -> bf16 (round-to-nearest-even), with zero-fill of padded rows.
// K is fixed at 256 elements per row.
// ---------------------------------------------------------------------------
__device__ __forceinline__ unsigned short f32_to_bf16_rne(float f) {
    unsigned int u = __float_as_uint(f);
    unsigned int r = u + 0x7FFFu + ((u >> 16) & 1u);
    return (unsigned short)(r >> 16);
}

__global__ void k_f32_to_bf16(const float* __restrict__ in,
                              unsigned short* __restrict__ out,
                              int rows) {
    int gid = blockIdx.x * 256 + threadIdx.x;   // rowsPad * 256 threads total
    int row = gid >> 8;
    float v = (row < rows) ? in[gid] : 0.0f;
    out[gid] = f32_to_bf16_rne(v);
}

// ---------------------------------------------------------------------------
// WMMA GEMM: C[M x NOUT] = A[M x 256](bf16) * B[NOUT x 256](bf16)^T (+ bias)
// NOUT==256: block = 8 waves -> 64 rows x 256 cols (wave: 16 x 128, 8 tiles)
// NOUT==64 : block = 8 waves -> 128 rows x 64 cols (wave: 16 x 64, 4 tiles)
// VGPR layouts per CDNA5 ISA 7.12.2 (wave32).
// ---------------------------------------------------------------------------
template<int NOUT, bool BIAS>
__global__ void k_gemm_wmma(const unsigned short* __restrict__ Abf,
                            const unsigned short* __restrict__ Bbf,
                            const float* __restrict__ bias,
                            float* __restrict__ C,
                            int mstore) {
    const int lane = threadIdx.x & 31;
    const int wave = threadIdx.x >> 5;
    const int half = lane >> 4;     // K-chunk selector (ISA A/B frag layout)
    const int l16  = lane & 15;     // row (A) / col (B,C,D) within tile

    constexpr int NT = (NOUT == 256) ? 8 : 4;
    int row0, ncol0;
    if (NOUT == 256) { row0 = blockIdx.x * 64  + (wave & 3) * 16; ncol0 = (wave >> 2) * 128; }
    else             { row0 = blockIdx.x * 128 + wave * 16;       ncol0 = 0; }

    const bf16_t* A = (const bf16_t*)Abf;
    const bf16_t* B = (const bf16_t*)Bbf;

    f32x8 acc[NT];
#pragma unroll
    for (int t = 0; t < NT; ++t) acc[t] = (f32x8){0.f,0.f,0.f,0.f,0.f,0.f,0.f,0.f};

    // A: lane holds row (row0+l16), K chunks [8*half, 8*half+8) and +16.
    const bf16_t* arow = A + (size_t)(row0 + l16) * HID + half * 8;

#pragma unroll
    for (int k0 = 0; k0 < HID; k0 += 32) {
        bf16x8 alo = *(const bf16x8*)(arow + k0);
        bf16x8 ahi = *(const bf16x8*)(arow + k0 + 16);
        bf16x16 af = __builtin_shufflevector(alo, ahi,
                         0,1,2,3,4,5,6,7,8,9,10,11,12,13,14,15);
#pragma unroll
        for (int t = 0; t < NT; ++t) {
            int ncol = ncol0 + t * 16 + l16;
            // B col ncol == row ncol of W (row-major): K chunks [8*half, 8*half+8) and +16
            const bf16_t* bp = B + (size_t)ncol * HID + k0 + half * 8;
            bf16x8 blo = *(const bf16x8*)(bp);
            bf16x8 bhi = *(const bf16x8*)(bp + 16);
            bf16x16 bfm = __builtin_shufflevector(blo, bhi,
                         0,1,2,3,4,5,6,7,8,9,10,11,12,13,14,15);
            acc[t] = __builtin_amdgcn_wmma_f32_16x16x32_bf16(
                         false, af, false, bfm, (short)0, acc[t], false, false);
        }
    }

    const int rbase = row0 + half * 8;   // D layout: row = r + 8*(lane/16)
#pragma unroll
    for (int t = 0; t < NT; ++t) {
        int col = ncol0 + t * 16 + l16;
        float bv = BIAS ? bias[col] : 0.0f;
#pragma unroll
        for (int r = 0; r < 8; ++r) {
            int row = rbase + r;
            if (row < mstore) C[(size_t)row * NOUT + col] = acc[t][r] + bv;
        }
    }
}

// ---------------------------------------------------------------------------
// Per-node attention scores: s_src/s_dst[n,h] = <h[n,h,:], a_*[h,:]>
// ---------------------------------------------------------------------------
__global__ void k_scores(const float* __restrict__ h,
                         const float* __restrict__ a_src,
                         const float* __restrict__ a_dst,
                         float* __restrict__ ssrc,
                         float* __restrict__ sdst,
                         int N) {
    int gid = blockIdx.x * 256 + threadIdx.x;
    if (gid >= N * HEADS) return;
    int n = gid >> 2, hd = gid & 3;
    const float* hp = h + (size_t)n * HID + hd * HD;
    const float* as = a_src + hd * HD;
    const float* ad = a_dst + hd * HD;
    float s = 0.f, d = 0.f;
#pragma unroll 8
    for (int i = 0; i < HD; ++i) { float v = hp[i]; s += v * as[i]; d += v * ad[i]; }
    ssrc[gid] = s;
    sdst[gid] = d;
}

__global__ void k_zero(float* __restrict__ p, int n) {
    int gid = blockIdx.x * 256 + threadIdx.x;
    if (gid < n) p[gid] = 0.0f;
}

// Pass 1: segment max (values clamped at >= 0 per reference include_self max).
// For non-negative floats the IEEE bit pattern is monotone -> int atomicMax.
__global__ void k_edge_max(const int* __restrict__ src, const int* __restrict__ dst,
                           const float* __restrict__ ssrc, const float* __restrict__ sdst,
                           int* __restrict__ emax, int E) {
    int gid = blockIdx.x * 256 + threadIdx.x;
    if (gid >= E * HEADS) return;
    int e = gid >> 2, hd = gid & 3;
    int s = src[e], d = dst[e];
    float v = ssrc[s * HEADS + hd] + sdst[d * HEADS + hd];
    v = (v >= 0.f) ? v : 0.2f * v;              // leaky_relu
    if (v > 0.f) atomicMax(&emax[d * HEADS + hd], __float_as_int(v));
}

// Pass 2: exp(e - max) into scratch + segment sum via atomics.
__global__ void k_edge_exp(const int* __restrict__ src, const int* __restrict__ dst,
                           const float* __restrict__ ssrc, const float* __restrict__ sdst,
                           const int* __restrict__ emax, float* __restrict__ esum,
                           float* __restrict__ eexp, int E) {
    int gid = blockIdx.x * 256 + threadIdx.x;
    if (gid >= E * HEADS) return;
    int e = gid >> 2, hd = gid & 3;
    int s = src[e], d = dst[e];
    float v = ssrc[s * HEADS + hd] + sdst[d * HEADS + hd];
    v = (v >= 0.f) ? v : 0.2f * v;
    float m = __int_as_float(emax[d * HEADS + hd]);
    float ex = __expf(v - m);
    eexp[gid] = ex;
    atomicAdd(&esum[d * HEADS + hd], ex);
}

// Pass 3: scatter messages. 64 threads per edge (one per head-dim), 4 edges
// per 256-thread block. h (51 MB) is L2-resident on MI455X (192 MB L2), so
// gathers + f32 atomics stay on-die.
__global__ void k_edge_scatter(const int* __restrict__ src, const int* __restrict__ dst,
                               const float* __restrict__ h,
                               const float* __restrict__ eexp,
                               const float* __restrict__ esum,
                               float* __restrict__ hagg, int E) {
    int eid = blockIdx.x * 4 + (threadIdx.x >> 6);
    int dl  = threadIdx.x & 63;
    if (eid >= E) return;
    int s = src[eid], dn = dst[eid];
    const float* hp = h    + (size_t)s  * HID;
    float*       op = hagg + (size_t)dn * HID;
#pragma unroll
    for (int hd = 0; hd < HEADS; ++hd) {
        float alpha = eexp[eid * HEADS + hd] / fmaxf(esum[dn * HEADS + hd], 1e-9f);
        atomicAdd(&op[hd * HD + dl], hp[hd * HD + dl] * alpha);
    }
}

__global__ void k_bn_elu(float* __restrict__ h,
                         const float* __restrict__ g, const float* __restrict__ b,
                         const float* __restrict__ m, const float* __restrict__ v,
                         int N) {
    int gid = blockIdx.x * 256 + threadIdx.x;
    if (gid >= N * HID) return;
    int c = gid & 255;
    float x = h[gid];
    x = (x - m[c]) * rsqrtf(v[c] + BN_EPS) * g[c] + b[c];
    h[gid] = (x > 0.f) ? x : expm1f(x);
}

// ---------------------------------------------------------------------------
extern "C" void kernel_launch(void* const* d_in, const int* in_sizes, int n_in,
                              void* d_out, int out_size, void* d_ws, size_t ws_size,
                              hipStream_t stream) {
    (void)n_in; (void)out_size; (void)ws_size;

    const float* x      = (const float*)d_in[0];
    const int*   ei     = (const int*)  d_in[1];
    const float* W1     = (const float*)d_in[2];
    const float* a_src1 = (const float*)d_in[3];
    const float* a_dst1 = (const float*)d_in[4];
    const float* bn1g   = (const float*)d_in[5];
    const float* bn1b   = (const float*)d_in[6];
    const float* bn1m   = (const float*)d_in[7];
    const float* bn1v   = (const float*)d_in[8];
    const float* W2     = (const float*)d_in[9];
    const float* a_src2 = (const float*)d_in[10];
    const float* a_dst2 = (const float*)d_in[11];
    const float* bn2g   = (const float*)d_in[12];
    const float* bn2b   = (const float*)d_in[13];
    const float* bn2m   = (const float*)d_in[14];
    const float* bn2v   = (const float*)d_in[15];
    const float* Wc     = (const float*)d_in[16];
    const float* bc     = (const float*)d_in[17];

    const int N = in_sizes[0] / HID;
    const int E = in_sizes[1] / 2;
    const int* src = ei;
    const int* dst = ei + E;
    const int Npad = (N + 127) & ~127;          // multiple of 128 for both GEMMs

    char* ws = (char*)d_ws;
    size_t off = 0;
    auto carve = [&](size_t bytes) -> char* {
        char* p = ws + off;
        off = (off + bytes + 255) & ~(size_t)255;
        return p;
    };
    unsigned short* xb   = (unsigned short*)carve((size_t)Npad * HID * 2);
    unsigned short* w1b  = (unsigned short*)carve((size_t)HID * HID * 2);
    unsigned short* w2b  = (unsigned short*)carve((size_t)HID * HID * 2);
    unsigned short* wcb  = (unsigned short*)carve((size_t)64 * HID * 2);
    float*          h    = (float*)carve((size_t)Npad * HID * 4);
    float*          hagg = (float*)carve((size_t)Npad * HID * 4);
    float*          ssrc = (float*)carve((size_t)N * HEADS * 4);
    float*          sdst = (float*)carve((size_t)N * HEADS * 4);
    int*            emax = (int*)  carve((size_t)N * HEADS * 4);
    float*          esum = (float*)carve((size_t)N * HEADS * 4);
    float*          eexp = (float*)carve((size_t)E * HEADS * 4);

    const int gNH   = (N * HEADS + 255) / 256;
    const int gEH   = (E * HEADS + 255) / 256;
    const int gFeat = Npad;                      // Npad*256 elems / 256 thr

    // Weight conversion (once per call; rows already aligned).
    k_f32_to_bf16<<<HID, 256, 0, stream>>>(W1, w1b, HID);
    k_f32_to_bf16<<<HID, 256, 0, stream>>>(W2, w2b, HID);
    k_f32_to_bf16<<<64,  256, 0, stream>>>(Wc, wcb, 64);

    // -------- Layer 1 --------
    k_f32_to_bf16<<<gFeat, 256, 0, stream>>>(x, xb, N);
    k_gemm_wmma<256, false><<<Npad / 64, 256, 0, stream>>>(xb, w1b, nullptr, h, Npad);
    k_scores<<<gNH, 256, 0, stream>>>(h, a_src1, a_dst1, ssrc, sdst, N);
    k_zero<<<gNH, 256, 0, stream>>>((float*)emax, N * HEADS);
    k_zero<<<gNH, 256, 0, stream>>>(esum, N * HEADS);
    k_zero<<<gFeat, 256, 0, stream>>>(hagg, Npad * HID);
    k_edge_max    <<<gEH, 256, 0, stream>>>(src, dst, ssrc, sdst, emax, E);
    k_edge_exp    <<<gEH, 256, 0, stream>>>(src, dst, ssrc, sdst, emax, esum, eexp, E);
    k_edge_scatter<<<(E + 3) / 4, 256, 0, stream>>>(src, dst, h, eexp, esum, hagg, E);
    k_bn_elu<<<N, 256, 0, stream>>>(hagg, bn1g, bn1b, bn1m, bn1v, N);

    // -------- Layer 2 --------
    k_f32_to_bf16<<<gFeat, 256, 0, stream>>>(hagg, xb, N);
    k_gemm_wmma<256, false><<<Npad / 64, 256, 0, stream>>>(xb, w2b, nullptr, h, Npad);
    k_scores<<<gNH, 256, 0, stream>>>(h, a_src2, a_dst2, ssrc, sdst, N);
    k_zero<<<gNH, 256, 0, stream>>>((float*)emax, N * HEADS);
    k_zero<<<gNH, 256, 0, stream>>>(esum, N * HEADS);
    k_zero<<<gFeat, 256, 0, stream>>>(hagg, Npad * HID);
    k_edge_max    <<<gEH, 256, 0, stream>>>(src, dst, ssrc, sdst, emax, E);
    k_edge_exp    <<<gEH, 256, 0, stream>>>(src, dst, ssrc, sdst, emax, esum, eexp, E);
    k_edge_scatter<<<(E + 3) / 4, 256, 0, stream>>>(src, dst, h, eexp, esum, hagg, E);
    k_bn_elu<<<N, 256, 0, stream>>>(hagg, bn2g, bn2b, bn2m, bn2v, N);

    // -------- Classifier --------
    k_f32_to_bf16<<<gFeat, 256, 0, stream>>>(hagg, xb, N);
    k_gemm_wmma<64, true><<<Npad / 128, 256, 0, stream>>>(xb, wcb, bc, (float*)d_out, N);
}
